// TensorMemory_34608846471713
// MI455X (gfx1250) — matrix operand order repaired and checked by reading, not compiled, not run on hardware
//
#include <hip/hip_runtime.h>

// Problem: B=2, H=32, S=4096, D=128
#define BB 2
#define HH 32
#define SS 4096
#define DD 128
#define BH (BB * HH)            // 64
#define EPS 1e-6f

#define SPLIT 4                  // S split for phase-1 occupancy (256 blocks)
#define SCHUNK (SS / SPLIT)      // 1024 rows per split
#define KTILE 32                 // matches WMMA K for bf16
#define P1_STRIDE 130            // f32 LDS row stride: conflict-free A & B frag reads
#define P2_SB 134                // bf16 LDS row stride for M in phase 2

typedef __bf16 bf16;
typedef __attribute__((ext_vector_type(16))) __bf16 v16bf;
typedef __attribute__((ext_vector_type(2)))  __bf16 v2bf;
typedef __attribute__((ext_vector_type(8)))  float  v8f;
typedef __attribute__((ext_vector_type(4)))  float  v4f;
typedef __attribute__((ext_vector_type(2)))  float  v2f;

#define AS1 __attribute__((address_space(1)))
#define AS3 __attribute__((address_space(3)))

// Pointee type exactly as the builtin's prototype prints it:
// 'int __attribute__((__vector_size__(2 * sizeof(int))))'
typedef int gi2 __attribute__((vector_size(8)));

#if defined(__has_builtin)
#if __has_builtin(__builtin_amdgcn_global_load_async_to_lds_b64)
#define USE_ASYNC_LDS 1
#endif
#endif

__device__ __forceinline__ void wait_async0() {
#if defined(__has_builtin)
#if __has_builtin(__builtin_amdgcn_s_wait_asynccnt)
  __builtin_amdgcn_s_wait_asynccnt(0);
  return;
#else
  asm volatile("s_wait_asynccnt 0" ::: "memory");
  return;
#endif
#else
  asm volatile("s_wait_asynccnt 0" ::: "memory");
#endif
}

#define MPART_FLOATS ((size_t)SPLIT * BH * DD * DD)   // 16 MB of f32 partial M

// Build a 32x16 bf16 B-fragment from an f32 LDS tile at column e0.
// Layout: lanes 0-15: v0..3 -> K=0..7, v4..7 -> K=16..23; lanes 16-31: +8.
__device__ __forceinline__ v16bf bfrag_f32(const float* lds, int e0, int half, int m16) {
  v16bf b;
#pragma unroll
  for (int v = 0; v < 8; ++v) {
    int k = (v < 4) ? (half * 8 + 2 * v) : (16 + half * 8 + 2 * (v - 4));
    b[2 * v]     = (bf16)lds[k       * P1_STRIDE + e0 + m16];
    b[2 * v + 1] = (bf16)lds[(k + 1) * P1_STRIDE + e0 + m16];
  }
  return b;
}

// ---------------------------------------------------------------------------
// Phase 1: per (bh, split) block computes partial  M = V^T K  (128x128 f32)
// and partial z = sum_s k_s over its S-chunk. 8 waves, wave w owns d-rows
// [16w, 16w+16), computes all 8 e-tiles -> 8 v_wmma per 32-row K-step.
// Tiles staged HBM -> LDS via GLOBAL_LOAD_ASYNC_TO_LDS_B64 (ASYNCcnt) when
// the toolchain exposes the builtin; synchronous copy otherwise.
// ---------------------------------------------------------------------------
__global__ __launch_bounds__(256) void tm_phase1(
    const float* __restrict__ Kg, const float* __restrict__ Vg,
    float* __restrict__ Mpart, float* __restrict__ zpart)
{
  __shared__ float ldsK[KTILE * P1_STRIDE];
  __shared__ float ldsV[KTILE * P1_STRIDE];

  const int bh    = blockIdx.x >> 2;
  const int split = blockIdx.x & 3;
  const int tid   = threadIdx.x;
  const int wave  = tid >> 5;
  const int lane  = tid & 31;
  const int m16   = lane & 15;
  const int half  = lane >> 4;
  const int d0    = wave * 16;

  const float* Kbh = Kg + (size_t)bh * SS * DD + (size_t)split * SCHUNK * DD;
  const float* Vbh = Vg + (size_t)bh * SS * DD + (size_t)split * SCHUNK * DD;

  v8f acc[8];
#pragma unroll
  for (int t = 0; t < 8; ++t)
#pragma unroll
    for (int e = 0; e < 8; ++e) acc[t][e] = 0.0f;

  const int ze = tid & 127;      // e-column this thread reduces for z
  const int zh = tid >> 7;       // which 16-row half of the K tile
  float zacc = 0.0f;

  for (int s0 = 0; s0 < SCHUNK; s0 += KTILE) {
    if (s0 + KTILE < SCHUNK) {   // gfx1250 global_prefetch_b8 of next tiles
      __builtin_prefetch(Kbh + (size_t)(s0 + KTILE) * DD + tid * 16, 0, 1);
      __builtin_prefetch(Vbh + (size_t)(s0 + KTILE) * DD + tid * 16, 0, 1);
    }
    __syncthreads();             // previous tile fully consumed

#if USE_ASYNC_LDS
    // Async DMA: 2048 b64 beats per 32x128 f32 tile, 8 per thread per tile.
    // Padded rows (520 B) are 8-byte aligned, so b64 is the legal beat size.
#pragma unroll
    for (int i = 0; i < 8; ++i) {
      int f   = tid + i * 256;       // float2 id, 0..2047
      int row = f >> 6;              // 64 float2 per 128-wide row
      int c   = (f & 63) << 1;       // float column
      __builtin_amdgcn_global_load_async_to_lds_b64(
          (AS1 gi2*)(Kbh + (size_t)s0 * DD + 2 * (size_t)f),
          (AS3 gi2*)&ldsK[row * P1_STRIDE + c], 0, 0);
      __builtin_amdgcn_global_load_async_to_lds_b64(
          (AS1 gi2*)(Vbh + (size_t)s0 * DD + 2 * (size_t)f),
          (AS3 gi2*)&ldsV[row * P1_STRIDE + c], 0, 0);
    }
    wait_async0();
#else
    const v4f* K4 = (const v4f*)(Kbh + (size_t)s0 * DD);
    const v4f* V4 = (const v4f*)(Vbh + (size_t)s0 * DD);
#pragma unroll
    for (int i = 0; i < 4; ++i) {
      int f   = tid + i * 256;       // float4 id, 0..1023
      int row = f >> 5;              // 32 float4 per 128-wide row
      int c   = (f & 31) << 2;       // float column
      v4f k = K4[f], v = V4[f];
      v2f klo = {k[0], k[1]}, khi = {k[2], k[3]};
      v2f vlo = {v[0], v[1]}, vhi = {v[2], v[3]};
      *(v2f*)&ldsK[row * P1_STRIDE + c]     = klo;
      *(v2f*)&ldsK[row * P1_STRIDE + c + 2] = khi;
      *(v2f*)&ldsV[row * P1_STRIDE + c]     = vlo;
      *(v2f*)&ldsV[row * P1_STRIDE + c + 2] = vhi;
    }
#endif
    __syncthreads();

    // z partial: K column sums (f32, exact)
#pragma unroll
    for (int r = 0; r < 16; ++r)
      zacc += ldsK[(zh * 16 + r) * P1_STRIDE + ze];

    // A fragment = V^T tile (A[m,k] = V[s0+k, d0+m]), 16x32 bf16 layout:
    // lanes 0-15: v0..3 -> K=0..7, v4..7 -> K=16..23; lanes 16-31: +8.
    v16bf a;
#pragma unroll
    for (int v = 0; v < 8; ++v) {
      int k = (v < 4) ? (half * 8 + 2 * v) : (16 + half * 8 + 2 * (v - 4));
      a[2 * v]     = (bf16)ldsV[k       * P1_STRIDE + d0 + m16];
      a[2 * v + 1] = (bf16)ldsV[(k + 1) * P1_STRIDE + d0 + m16];
    }

    // Double-buffered B fragments: avoids the WMMA->VALU WAR hazard NOPs
    // (next fragment's cvt writes a different register set).
#pragma unroll
    for (int et = 0; et < 8; et += 2) {
      v16bf b0 = bfrag_f32(ldsK, et * 16, half, m16);
      v16bf b1 = bfrag_f32(ldsK, et * 16 + 16, half, m16);
      acc[et] = __builtin_amdgcn_wmma_f32_16x16x32_bf16(
          false, a, false, b0, (short)0, acc[et], false, false);
      acc[et + 1] = __builtin_amdgcn_wmma_f32_16x16x32_bf16(
          false, a, false, b1, (short)0, acc[et + 1], false, false);
    }
  }

  // Write partial M (C/D layout: lane n = lane&15, VGPR r -> m = r + 8*half)
  float* Mo = Mpart + ((size_t)split * BH + bh) * (DD * DD);
#pragma unroll
  for (int et = 0; et < 8; ++et)
#pragma unroll
    for (int r = 0; r < 8; ++r) {
      int row = d0 + r + half * 8;
      int col = et * 16 + m16;
      Mo[row * DD + col] = acc[et][r];
    }

  // Partial z: combine the two 16-row halves through LDS
  __syncthreads();
  ldsK[zh * 128 + ze] = zacc;
  __syncthreads();
  if (tid < 128)
    zpart[((size_t)split * BH + bh) * DD + tid] = ldsK[tid] + ldsK[128 + tid];
}

// B-fragment for phase 2 from the bf16 M tile in LDS.
__device__ __forceinline__ v16bf bfrag_bf16(const bf16* Ml, int d0, int e0,
                                            int half, int m16) {
  v16bf b;
#pragma unroll
  for (int v = 0; v < 8; ++v) {
    v2bf mm = *(const v2bf*)&Ml[(d0 + m16) * P2_SB + e0 + ((v < 4) ? (half * 8 + 2 * v) : (16 + half * 8 + 2 * (v - 4)))];
    b[2 * v] = mm[0]; b[2 * v + 1] = mm[1];
  }
  return b;
}

// ---------------------------------------------------------------------------
// Phase 2: out[s,d] = (sum_e M[d,e] Q[s,e]) / (z.q_s + eps).
// Each block: one (bh, 128-row s-tile). M partials (17 MB total) are re-read
// from L2 (192 MB) and summed while staging M into LDS as bf16.
// Wave w computes a 16x128 output tile: 4 K-chunks x 8 d-tiles = 32 v_wmma.
// ---------------------------------------------------------------------------
__global__ __launch_bounds__(256) void tm_phase2(
    const float* __restrict__ Qg, const float* __restrict__ Mpart,
    const float* __restrict__ zpart, float* __restrict__ Out)
{
  __shared__ bf16  Ml[DD * P2_SB];
  __shared__ float zl[DD];

  const int bh    = blockIdx.x >> 5;   // 0..63
  const int stile = blockIdx.x & 31;   // 0..31
  const int tid   = threadIdx.x;
  const int wave  = tid >> 5;
  const int lane  = tid & 31;
  const int m16   = lane & 15;
  const int half  = lane >> 4;

  // Stage M = sum of SPLIT partials (L2-resident), convert once to bf16.
  const float* P0 = Mpart + ((size_t)0 * BH + bh) * (DD * DD);
  const float* P1 = Mpart + ((size_t)1 * BH + bh) * (DD * DD);
  const float* P2 = Mpart + ((size_t)2 * BH + bh) * (DD * DD);
  const float* P3 = Mpart + ((size_t)3 * BH + bh) * (DD * DD);
  for (int f = tid; f < DD * DD / 4; f += 256) {
    v4f t = ((const v4f*)P0)[f] + ((const v4f*)P1)[f] +
            ((const v4f*)P2)[f] + ((const v4f*)P3)[f];
    int row = f >> 5;
    int c   = (f & 31) << 2;
    v2bf lo = {(bf16)t[0], (bf16)t[1]};
    v2bf hi = {(bf16)t[2], (bf16)t[3]};
    *(v2bf*)&Ml[row * P2_SB + c]     = lo;
    *(v2bf*)&Ml[row * P2_SB + c + 2] = hi;
  }
  if (tid < DD)
    zl[tid] = zpart[(0 * BH + bh) * DD + tid] + zpart[(1 * BH + bh) * DD + tid] +
              zpart[(2 * BH + bh) * DD + tid] + zpart[(3 * BH + bh) * DD + tid];
  __syncthreads();

  const int sBase = stile * 128 + wave * 16;
  const int srow  = sBase + m16;                 // A layout: both halves, m=lane&15
  const v4f* q4 = (const v4f*)(Qg + ((size_t)bh * SS + srow) * DD);

  v8f acc[8];
#pragma unroll
  for (int t = 0; t < 8; ++t)
#pragma unroll
    for (int e = 0; e < 8; ++e) acc[t][e] = 0.0f;

  float zdot = 0.0f;   // this lane's disjoint half of z . q_srow (f32 exact)

#pragma unroll
  for (int c = 0; c < 4; ++c) {
    const int e0 = c * 32;
    const int eb0 = e0 + half * 8;
    const int eb1 = e0 + 16 + half * 8;
    v4f q0 = q4[eb0 >> 2];
    v4f q1 = q4[(eb0 >> 2) + 1];
    v4f q2 = q4[eb1 >> 2];
    v4f q3 = q4[(eb1 >> 2) + 1];

    v16bf a;
    a[0]  = (bf16)q0[0]; a[1]  = (bf16)q0[1]; a[2]  = (bf16)q0[2]; a[3]  = (bf16)q0[3];
    a[4]  = (bf16)q1[0]; a[5]  = (bf16)q1[1]; a[6]  = (bf16)q1[2]; a[7]  = (bf16)q1[3];
    a[8]  = (bf16)q2[0]; a[9]  = (bf16)q2[1]; a[10] = (bf16)q2[2]; a[11] = (bf16)q2[3];
    a[12] = (bf16)q3[0]; a[13] = (bf16)q3[1]; a[14] = (bf16)q3[2]; a[15] = (bf16)q3[3];

    zdot += zl[eb0+0]*q0[0] + zl[eb0+1]*q0[1] + zl[eb0+2]*q0[2] + zl[eb0+3]*q0[3]
          + zl[eb0+4]*q1[0] + zl[eb0+5]*q1[1] + zl[eb0+6]*q1[2] + zl[eb0+7]*q1[3]
          + zl[eb1+0]*q2[0] + zl[eb1+1]*q2[1] + zl[eb1+2]*q2[2] + zl[eb1+3]*q2[3]
          + zl[eb1+4]*q3[0] + zl[eb1+5]*q3[1] + zl[eb1+6]*q3[2] + zl[eb1+7]*q3[3];

#pragma unroll
    for (int dt = 0; dt < 8; dt += 2) {
      v16bf b0 = bfrag_bf16(Ml, dt * 16, e0, half, m16);
      v16bf b1 = bfrag_bf16(Ml, dt * 16 + 16, e0, half, m16);
      acc[dt] = __builtin_amdgcn_wmma_f32_16x16x32_bf16(
          false, a, false, b0, (short)0, acc[dt], false, false);
      acc[dt + 1] = __builtin_amdgcn_wmma_f32_16x16x32_bf16(
          false, a, false, b1, (short)0, acc[dt + 1], false, false);
    }
  }

  // Denominator: lanes l and l^16 cover disjoint halves of the dot product.
  float denom = zdot + __shfl_xor(zdot, 16, 32) + EPS;
  float inv = 1.0f / denom;

  float* obh = Out + (size_t)bh * SS * DD;
#pragma unroll
  for (int r = 0; r < 8; ++r) {
    int m = r + half * 8;                    // output row within the 16-tile
    float invm = __shfl(inv, m, 32);         // denom of that row (lane m holds it)
    float* orow = obh + (size_t)(sBase + m) * DD;
#pragma unroll
    for (int dt = 0; dt < 8; ++dt)
      orow[dt * 16 + m16] = acc[dt][r] * invm;
  }
}

// ---------------------------------------------------------------------------
extern "C" void kernel_launch(void* const* d_in, const int* in_sizes, int n_in,
                              void* d_out, int out_size, void* d_ws, size_t ws_size,
                              hipStream_t stream) {
  const float* Kg = (const float*)d_in[0];   // keys    [B,H,S,D]
  const float* Vg = (const float*)d_in[1];   // values  [B,H,S,D]
  const float* Qg = (const float*)d_in[2];   // queries [B,H,S,D]
  float* Out = (float*)d_out;                // [B,H,S,D] f32

  // Workspace: SPLIT partial M (16 MB) + SPLIT partial z (128 KB).
  float* Mpart = (float*)d_ws;
  float* zpart = Mpart + MPART_FLOATS;

  tm_phase1<<<BH * SPLIT, 256, 0, stream>>>(Kg, Vg, Mpart, zpart);
  tm_phase2<<<BH * (SS / 128), 256, 0, stream>>>(Qg, Mpart, zpart, Out);
}
